// NeuralNetwork_32641751449680
// MI455X (gfx1250) — matrix operand order repaired and checked by reading, not compiled, not run on hardware
//
#include <hip/hip_runtime.h>
#include <hip/hip_bf16.h>

typedef __attribute__((ext_vector_type(16))) _Float16 v16h;
typedef __attribute__((ext_vector_type(8)))  _Float16 v8h;
typedef __attribute__((ext_vector_type(8)))  float    v8f;

#define NN_B 16384
#define NN_S 510
#define NN_D 1024

// ---------------------------------------------------------------- helpers ---

__global__ __launch_bounds__(256) void f32_to_f16_kernel(
    const float* __restrict__ s, _Float16* __restrict__ d, int n) {
  int i = blockIdx.x * 256 + threadIdx.x;
  if (i < n) d[i] = (_Float16)s[i];
}

__global__ __launch_bounds__(256) void build_ctx_kernel(
    const float* __restrict__ inp, const float* __restrict__ task,
    _Float16* __restrict__ ctx16) {
  int i = blockIdx.x * 256 + threadIdx.x;            // over B*1024
  int col = i & (NN_D - 1);
  int row = i >> 10;
  float v = (col < NN_S) ? inp[(size_t)row * NN_S + col]
                         : task[(size_t)row * (NN_S + 4) + (col - NN_S)];
  ctx16[i] = (_Float16)v;
}

__global__ __launch_bounds__(256) void bias_sum_kernel(
    const float* __restrict__ a, const float* __restrict__ b,
    float* __restrict__ c, int n) {
  int i = blockIdx.x * 256 + threadIdx.x;
  if (i < n) c[i] = a[i] + b[i];
}

// -------------------------------------------------------- WMMA f16 GEMM -----
// C[M,N] = epilogue( A[M,K] @ W[N,K]^T )   (all row-major; f16 in, f32 acc)
// Wave computes a 16x64 tile: 1 A fragment reused across 4 WMMAs per K-step.
// Workgroup = 8 waves stacked in M -> 128(M) x 64(N) per block.

__global__ __launch_bounds__(256) void gemm_wmma_f16(
    const _Float16* __restrict__ A,    // [M,K] f16
    const _Float16* __restrict__ W,    // [N,K] f16  (B = W^T)
    const float*    __restrict__ bias, // [N] or null
    const float*    __restrict__ Cin,  // [M,N] f32 or null (added)
    float*          __restrict__ C32,  // [M,N] f32 or null
    _Float16*       __restrict__ C16,  // [M,N] f16 or null
    int M, int N, int K, int act)      // act: 0=none, 1=tanh
{
  const int tid  = threadIdx.x;
  const int lane = tid & 31;
  const int wave = tid >> 5;
  const int g    = lane >> 4;   // half-wave group (0/1)
  const int l16  = lane & 15;

  const int m0 = blockIdx.y * 128 + wave * 16;
  const int n0 = blockIdx.x * 64;

  const _Float16* Arow = A + (size_t)(m0 + l16) * K;
  const _Float16* W0 = W + (size_t)(n0 +  0 + l16) * K + g * 16;
  const _Float16* W1 = W + (size_t)(n0 + 16 + l16) * K + g * 16;
  const _Float16* W2 = W + (size_t)(n0 + 32 + l16) * K + g * 16;
  const _Float16* W3 = W + (size_t)(n0 + 48 + l16) * K + g * 16;

  v8f c0 = {}, c1 = {}, c2 = {}, c3 = {};

  for (int k0 = 0; k0 < K; k0 += 32) {
    // A fragment: lane holds row m0+l16, K = g*16 + e (contiguous)
    v8h alo = *(const v8h*)(Arow + k0 + g * 16);
    v8h ahi = *(const v8h*)(Arow + k0 + g * 16 + 8);
    v16h a;
#pragma unroll
    for (int e = 0; e < 8; ++e) { a[e] = alo[e]; a[e + 8] = ahi[e]; }

    // B fragments: lane holds column n0+t*16+l16, K = g*16 + e (contiguous)
    v16h b0 = *(const v16h*)(W0 + k0);
    v16h b1 = *(const v16h*)(W1 + k0);
    v16h b2 = *(const v16h*)(W2 + k0);
    v16h b3 = *(const v16h*)(W3 + k0);

    c0 = __builtin_amdgcn_wmma_f32_16x16x32_f16(false, a, false, b0, (short)0, c0, false, false);
    c1 = __builtin_amdgcn_wmma_f32_16x16x32_f16(false, a, false, b1, (short)0, c1, false, false);
    c2 = __builtin_amdgcn_wmma_f32_16x16x32_f16(false, a, false, b2, (short)0, c2, false, false);
    c3 = __builtin_amdgcn_wmma_f32_16x16x32_f16(false, a, false, b3, (short)0, c3, false, false);
  }

  // C/D layout: VGPR r -> M = g*8 + r, N = l16
  const int crow = m0 + g * 8;
  v8f acc[4] = {c0, c1, c2, c3};
#pragma unroll
  for (int t = 0; t < 4; ++t) {
    const int ccol = n0 + t * 16 + l16;
    const float bb = bias ? bias[ccol] : 0.0f;
#pragma unroll
    for (int r = 0; r < 8; ++r) {
      size_t off = (size_t)(crow + r) * N + ccol;
      float v = acc[t][r] + bb;
      if (Cin) v += Cin[off];
      if (act) v = tanhf(v);
      if (C32) C32[off] = v;
      if (C16) C16[off] = (_Float16)v;
    }
  }
}

// ----------------------------------------------------- per-row argmax -------
// softmax is monotone -> argmax of logits == argmax of softmax. One wave/row.

__global__ __launch_bounds__(256) void argmax_rows(
    const float* __restrict__ X, int* __restrict__ kout, int N) {
  int row  = blockIdx.x * 8 + (threadIdx.x >> 5);
  int lane = threadIdx.x & 31;
  const float* xr = X + (size_t)row * N;
  float best = -__builtin_inff();
  int   bi   = 0x7FFFFFFF;
  for (int j = lane; j < N; j += 32) {
    float v = xr[j];
    if (v > best) { best = v; bi = j; }   // first occurrence within lane
  }
  for (int off = 16; off > 0; off >>= 1) {
    float ov = __shfl_xor(best, off, 32);
    int   oi = __shfl_xor(bi,   off, 32);
    if (ov > best || (ov == best && oi < bi)) { best = ov; bi = oi; }
  }
  if (lane == 0) kout[row] = bi;
}

// --------------------------------------------------------------- kwta -------
// thr = k-th largest value of the row (k = kidx[row]); keep x>=thr, else x/3.
// Radix select (4x 8-bit passes) on order-preserving uint encoding.

__device__ inline unsigned f2ord(float f) {
  unsigned u = __float_as_uint(f);
  return (u & 0x80000000u) ? ~u : (u | 0x80000000u);
}

__global__ __launch_bounds__(256) void kwta_kernel(
    float* __restrict__ X, _Float16* __restrict__ X16,
    const int* __restrict__ kidx, int N) {
  const int row = blockIdx.x;
  const int tid = threadIdx.x;
  float*    xr   = X   + (size_t)row * N;
  _Float16* xr16 = X16 + (size_t)row * N;
  const int k  = kidx[row];
  const int ne = N >> 8;                 // elements per thread (2 or 4)

  float    xv[4];
  unsigned uv[4];
#pragma unroll
  for (int e = 0; e < 4; ++e) {
    if (e < ne) {
      int j = tid + (e << 8);
      float v = xr[j];
      xv[e] = v;
      uv[e] = f2ord(v);
    }
  }

  __shared__ unsigned hist[256];
  __shared__ unsigned s_prefix;
  __shared__ int      s_k;

  unsigned thr_ord = 0u;
  if (k > 0) {
    if (tid == 0) { s_prefix = 0u; s_k = k; }
    __syncthreads();
    for (int pass = 0; pass < 4; ++pass) {
      const int shift = 24 - 8 * pass;
      hist[tid] = 0u;
      __syncthreads();
      unsigned prefix = s_prefix;
      unsigned himask = 0u;
      if (pass) himask = 0xFFFFFFFFu << (shift + 8);
      for (int e = 0; e < ne; ++e) {
        unsigned u = uv[e];
        if ((u & himask) == prefix)
          atomicAdd(&hist[(u >> shift) & 0xFFu], 1u);
      }
      __syncthreads();
      if (tid == 0) {
        unsigned cum = 0;
        int kk = s_k;
        for (int b = 255; b >= 0; --b) {
          unsigned c = hist[b];
          if (cum + c >= (unsigned)kk) {
            s_k = kk - (int)cum;
            s_prefix = prefix | ((unsigned)b << shift);
            break;
          }
          cum += c;
        }
      }
      __syncthreads();
    }
    thr_ord = s_prefix;
    __syncthreads();
  }

  for (int e = 0; e < ne; ++e) {
    int j = tid + (e << 8);
    float v = xv[e];
    float y = (k > 0 && uv[e] >= thr_ord) ? v : v / 3.0f;
    xr[j]   = y;
    xr16[j] = (_Float16)y;
  }
}

// ------------------------------------------------------------- launcher -----

extern "C" void kernel_launch(void* const* d_in, const int* in_sizes, int n_in,
                              void* d_out, int out_size, void* d_ws, size_t ws_size,
                              hipStream_t stream) {
  const float* input  = (const float*)d_in[0];
  const float* task   = (const float*)d_in[1];
  const float* hidden = (const float*)d_in[2];
  const float* w_ih   = (const float*)d_in[3];  const float* b_ih   = (const float*)d_in[4];
  const float* w_hh   = (const float*)d_in[5];  const float* b_hh   = (const float*)d_in[6];
  const float* w_cx11 = (const float*)d_in[7];  const float* b_cx11 = (const float*)d_in[8];
  const float* w_cx12 = (const float*)d_in[9];  const float* b_cx12 = (const float*)d_in[10];
  const float* w_cx21 = (const float*)d_in[11]; const float* b_cx21 = (const float*)d_in[12];
  const float* w_cx22 = (const float*)d_in[13]; const float* b_cx22 = (const float*)d_in[14];
  const float* w_cx31 = (const float*)d_in[15]; const float* b_cx31 = (const float*)d_in[16];
  const float* w_cx32 = (const float*)d_in[17]; const float* b_cx32 = (const float*)d_in[18];
  const float* w_l1   = (const float*)d_in[19]; const float* b_l1   = (const float*)d_in[20];
  const float* w_l2   = (const float*)d_in[21]; const float* b_l2   = (const float*)d_in[22];
  const float* w_l3   = (const float*)d_in[23]; const float* b_l3   = (const float*)d_in[24];
  const float* w_l4   = (const float*)d_in[25]; const float* b_l4   = (const float*)d_in[26];

  char* ws = (char*)d_ws;
  size_t off = 0;
  auto alloc = [&](size_t bytes) -> char* {
    off = (off + 255) & ~(size_t)255;
    char* p = ws + off;
    off += bytes;
    return p;
  };
  auto cvt = [&](const float* src, int n) -> _Float16* {
    _Float16* d = (_Float16*)alloc((size_t)n * sizeof(_Float16));
    f32_to_f16_kernel<<<(n + 255) / 256, 256, 0, stream>>>(src, d, n);
    return d;
  };

  // f16 weight copies (~20 MB total, L2 resident)
  _Float16* ih16  = cvt(w_ih,   1024 * 1024);
  _Float16* hh16  = cvt(w_hh,   1024 * 1024);
  _Float16* c1116 = cvt(w_cx11, 1024 * 1024);
  _Float16* c1216 = cvt(w_cx12, 1024 * 1024);
  _Float16* c2116 = cvt(w_cx21,  512 * 1024);
  _Float16* c2216 = cvt(w_cx22,  512 * 512);
  _Float16* c3116 = cvt(w_cx31, 1024 * 1024);
  _Float16* c3216 = cvt(w_cx32, 1024 * 1024);
  _Float16* l116  = cvt(w_l1,   1024 * 1024);
  _Float16* l216  = cvt(w_l2,    512 * 1024);
  _Float16* l316  = cvt(w_l3,   1024 * 512);
  _Float16* l416  = cvt(w_l4,   1024 * 1024);

  float* bsum = (float*)alloc(1024 * sizeof(float));
  bias_sum_kernel<<<4, 256, 0, stream>>>(b_ih, b_hh, bsum, 1024);

  const size_t BD = (size_t)NN_B * NN_D;
  _Float16* ctx16 = (_Float16*)alloc(BD * 2);
  _Float16* h016  = (_Float16*)alloc(BD * 2);
  _Float16* out16 = (_Float16*)alloc(BD * 2);
  _Float16* act16 = (_Float16*)alloc(BD * 2);
  float*    buf32 = (float*)alloc(BD * 4);
  int* k1 = (int*)alloc(NN_B * 4);
  int* k2 = (int*)alloc(NN_B * 4);
  int* k3 = (int*)alloc(NN_B * 4);

  build_ctx_kernel<<<(int)(BD / 256), 256, 0, stream>>>(input, task, ctx16);
  f32_to_f16_kernel<<<(int)((BD + 255) / 256), 256, 0, stream>>>(hidden, h016, (int)BD);

  float* out_x = (float*)d_out;          // final x [B,1024]
  float* out_h = out_x + BD;             // next_hidden [1,B,1024]

  auto gemm = [&](const _Float16* A, const _Float16* Wm, const float* bias,
                  const float* Cin, float* C32, _Float16* C16,
                  int M, int N, int K, int act) {
    dim3 grid(N / 64, M / 128);
    gemm_wmma_f16<<<grid, 256, 0, stream>>>(A, Wm, bias, Cin, C32, C16, M, N, K, act);
  };

  // out = tanh(ctx@W_ih^T + h0@W_hh^T + b_ih + b_hh); also = next_hidden
  gemm(ctx16, ih16, nullptr, nullptr, buf32, nullptr, NN_B, 1024, 1024, 0);
  gemm(h016,  hh16, bsum,    buf32,   out_h, out16,   NN_B, 1024, 1024, 1);

  // cx1 -> k1   (softmax skipped: argmax of logits)
  gemm(out16, c1116, b_cx11, nullptr, nullptr, act16, NN_B, 1024, 1024, 1);
  gemm(act16, c1216, b_cx12, nullptr, buf32,  nullptr, NN_B, 1024, 1024, 0);
  argmax_rows<<<NN_B / 8, 256, 0, stream>>>(buf32, k1, 1024);

  // cx2 -> k2
  gemm(out16, c2116, b_cx21, nullptr, nullptr, act16, NN_B, 512, 1024, 1);
  gemm(act16, c2216, b_cx22, nullptr, buf32,  nullptr, NN_B, 512, 512, 0);
  argmax_rows<<<NN_B / 8, 256, 0, stream>>>(buf32, k2, 512);

  // cx3 -> k3
  gemm(out16, c3116, b_cx31, nullptr, nullptr, act16, NN_B, 1024, 1024, 1);
  gemm(act16, c3216, b_cx32, nullptr, buf32,  nullptr, NN_B, 1024, 1024, 0);
  argmax_rows<<<NN_B / 8, 256, 0, stream>>>(buf32, k3, 1024);

  // main path: l1 -> kwta -> l2 -> kwta -> l3 -> kwta -> l4
  gemm(out16, l116, b_l1, nullptr, buf32, nullptr, NN_B, 1024, 1024, 0);
  kwta_kernel<<<NN_B, 256, 0, stream>>>(buf32, act16, k1, 1024);
  gemm(act16, l216, b_l2, nullptr, buf32, nullptr, NN_B, 512, 1024, 0);
  kwta_kernel<<<NN_B, 256, 0, stream>>>(buf32, act16, k2, 512);
  gemm(act16, l316, b_l3, nullptr, buf32, nullptr, NN_B, 1024, 512, 0);
  kwta_kernel<<<NN_B, 256, 0, stream>>>(buf32, act16, k3, 1024);
  gemm(act16, l416, b_l4, nullptr, out_x, nullptr, NN_B, 1024, 1024, 0);
}
